// MultiDimensionalRNNBase_23252952940717
// MI455X (gfx1250) — matrix operand order repaired and checked by reading, not compiled, not run on hardware
//
#include <hip/hip_runtime.h>
#include <math.h>

typedef _Float16 f16;
typedef _Float16 v16h __attribute__((ext_vector_type(16)));
typedef _Float16 v8h  __attribute__((ext_vector_type(8)));
typedef float    v8f  __attribute__((ext_vector_type(8)));

#define HID 64
#define HH  64      // image H (recurrence runs a 2-tap conv over this axis)
#define WW  256     // image W (sequential scan axis)
#define STRIDE 72   // f16 per LDS row: 144B rows -> 16B aligned, odd 16B-chunk (9 x 16B) bank spread

static __device__ __forceinline__ float hw_tanh(float v) {
#if __has_builtin(__builtin_amdgcn_tanhf)
    return __builtin_amdgcn_tanhf(v);          // v_tanh_f32 (gfx1250 TRANS op)
#elif __has_builtin(__builtin_amdgcn_tanh_f32)
    return __builtin_amdgcn_tanh_f32(v);
#else
    return tanhf(v);
#endif
}

// One block per (batch, direction) chain (32 chains total).
// Per scan step: new_state = tanh( WS[64x128] * SIN[128x64] + bias + x*w_in )
// where SIN rows 0..63  = state[:, h-1]  (SA image, zero-padded at h=-1)
//       SIN rows 64..127= state[:, h]    (SB image)
// State is ping-ponged between two parity copies so each step needs only ONE
// barrier (write->read); reads and writes never touch the same buffer.
// 16 waves, one 16x16 D-tile each; K=128 -> 4 x v_wmma_f32_16x16x32_f16 / wave / step.
__global__ __launch_bounds__(512)
void mdrnn_scan_kernel(const float* __restrict__ x,
                       const float* __restrict__ W_in,
                       const float* __restrict__ b_in,
                       const float* __restrict__ W_state,
                       const float* __restrict__ b_state,
                       float* __restrict__ out)
{
    // SBUF[parity][0] = SA (shifted state), SBUF[parity][1] = SB (state)
    __shared__ __attribute__((aligned(32))) f16 SBUF[2][2][HID * STRIDE];

    const int tid    = threadIdx.x;
    const int lane   = tid & 31;
    const int wave   = tid >> 5;        // 0..15
    const int o_tile = wave & 3;        // which 16 output channels
    const int j_tile = wave >> 2;       // which 16 h-rows

    const int n = blockIdx.x;           // 0..31
    const int b = n & 7;                // batch
    const int d = n >> 3;               // direction: 0=none 1=flipH 2=flipW 3=both
    const bool flipH = (d & 1) != 0;
    const bool flipW = (d & 2) != 0;

    // zero initial state (both parities; SA col 0 stays zero forever)
    for (int i = tid; i < 2 * 2 * HID * STRIDE; i += 512)
        (&SBUF[0][0][0])[i] = (f16)0;

    // ---- persistent A fragments (weights), ISA 16-bit A 16x32 layout:
    //      lane = M (o within tile); halves 0..7 -> K = (lane<16?0:8)+hh,
    //      halves 8..15 -> K = (lane<16?16:24)+(hh-8).  4 K-chunks of 32.
    // WS[o][k] = k<64 ? W_state[o][k][0] : W_state[o][k-64][1]
    const int m     = lane & 15;
    const int o_row = o_tile * 16 + m;
    v16h afrag[4];
#pragma unroll
    for (int c = 0; c < 4; ++c) {
#pragma unroll
        for (int hh = 0; hh < 16; ++hh) {
            int kloc = (hh < 8) ? (((lane < 16) ? 0 : 8) + hh)
                                : (((lane < 16) ? 16 : 24) + (hh - 8));
            int K = c * 32 + kloc;
            float wv = (K < HID) ? W_state[(o_row * HID + K) * 2 + 0]
                                 : W_state[(o_row * HID + (K - HID)) * 2 + 1];
            afrag[c][hh] = (f16)wv;
        }
    }

    // ---- per-lane epilogue constants for its 8 C/D elements:
    //      C layout: VGPR v -> M = v + (lane<16?0:8), N = lane%16
    const int mbase = (lane < 16) ? 0 : 8;
    const int o0    = o_tile * 16 + mbase;       // first of this lane's 8 output rows
    float win_r[8], bias_r[8];
#pragma unroll
    for (int v = 0; v < 8; ++v) {
        win_r[v]  = W_in[o0 + v];                // 1x1 conv, C==1
        bias_r[v] = b_in[o0 + v] + b_state[o0 + v];
    }

    const int h  = j_tile * 16 + (lane & 15);    // h-row this lane produces
    const int hp = flipH ? (HH - 1 - h) : h;

    // incremental pointers: no per-step flip/select arithmetic
    const int    dstep = flipW ? -1 : 1;
    const float* xp    = x + (size_t)(b * HH + hp) * WW + (flipW ? WW - 1 : 0);
    float*       outp  = out + ((size_t)((b * 4 + d) * HID + o0) * HH + h) * WW;

    // loop-invariant LDS fragment pointers per parity (LDS image is [h][channel]; lane%16 = N row, K along columns)
    const f16* pB[2][4];
    f16*       sb[2];
    f16*       sa[2];
#pragma unroll
    for (int p = 0; p < 2; ++p) {
        pB[p][0] = SBUF[p][0] + (j_tile * 16 + (lane & 15)) * STRIDE + (lane >> 4) * 8;       // K  0..31 (h-1)
        pB[p][1] = SBUF[p][0] + (j_tile * 16 + (lane & 15)) * STRIDE + (lane >> 4) * 8 + 32;  // K 32..63 (h-1)
        pB[p][2] = SBUF[p][1] + (j_tile * 16 + (lane & 15)) * STRIDE + (lane >> 4) * 8;       // K 64..95 (h)
        pB[p][3] = SBUF[p][1] + (j_tile * 16 + (lane & 15)) * STRIDE + (lane >> 4) * 8 + 32;  // K 96..127(h)
        sa[p] = SBUF[p][0] + (h + 1) * STRIDE + o0;
        sb[p] = SBUF[p][1] + h * STRIDE + o0;
    }
    const bool writeSA = (h < HH - 1);

    __syncthreads();

    for (int w = 0; w < WW; w += 2) {
#pragma unroll
        for (int par = 0; par < 2; ++par) {      // read parity=par, write parity=par^1
            const int wcur = w + par;

            // ---- issue all memory ops, then the wmma chain
            float xv = *xp;
            v8h l0a = *(const v8h*)(pB[par][0]);  v8h l0b = *(const v8h*)(pB[par][0] + 16);
            v8h l1a = *(const v8h*)(pB[par][1]);  v8h l1b = *(const v8h*)(pB[par][1] + 16);
            v8h l2a = *(const v8h*)(pB[par][2]);  v8h l2b = *(const v8h*)(pB[par][2] + 16);
            v8h l3a = *(const v8h*)(pB[par][3]);  v8h l3b = *(const v8h*)(pB[par][3] + 16);
            __builtin_prefetch(xp + dstep, 0, 1);    // global_prefetch_b8 next column

            v16h b0 = __builtin_shufflevector(l0a, l0b, 0,1,2,3,4,5,6,7,8,9,10,11,12,13,14,15);
            v16h b1 = __builtin_shufflevector(l1a, l1b, 0,1,2,3,4,5,6,7,8,9,10,11,12,13,14,15);
            v16h b2 = __builtin_shufflevector(l2a, l2b, 0,1,2,3,4,5,6,7,8,9,10,11,12,13,14,15);
            v16h b3 = __builtin_shufflevector(l3a, l3b, 0,1,2,3,4,5,6,7,8,9,10,11,12,13,14,15);

            v8f acc = __builtin_amdgcn_wmma_f32_16x16x32_f16(false, afrag[0], false, b0, (short)0, v8f{}, false, false);
            acc     = __builtin_amdgcn_wmma_f32_16x16x32_f16(false, afrag[1], false, b1, (short)0, acc,   false, false);
            acc     = __builtin_amdgcn_wmma_f32_16x16x32_f16(false, afrag[2], false, b2, (short)0, acc,   false, false);
            acc     = __builtin_amdgcn_wmma_f32_16x16x32_f16(false, afrag[3], false, b3, (short)0, acc,   false, false);

            float ns[8];
#pragma unroll
            for (int v = 0; v < 8; ++v) {
                float val = acc[v] + fmaf(xv, win_r[v], bias_r[v]);
                ns[v] = hw_tanh(val);                       // v_tanh_f32
                outp[(size_t)v * HH * WW + wcur] = ns[v];   // flipped-frame coords
            }
#pragma unroll
            for (int v = 0; v < 8; ++v)
                sb[par ^ 1][v] = (f16)ns[v];
            if (writeSA) {
#pragma unroll
                for (int v = 0; v < 8; ++v)
                    sa[par ^ 1][v] = (f16)ns[v];
            }

            xp += dstep;
            __syncthreads();   // single barrier: writes visible before next step's reads
        }
    }
}

extern "C" void kernel_launch(void* const* d_in, const int* in_sizes, int n_in,
                              void* d_out, int out_size, void* d_ws, size_t ws_size,
                              hipStream_t stream) {
    const float* x       = (const float*)d_in[0];
    const float* W_in    = (const float*)d_in[1];
    const float* b_in    = (const float*)d_in[2];
    const float* W_state = (const float*)d_in[3];
    const float* b_state = (const float*)d_in[4];
    float* out = (float*)d_out;
    (void)in_sizes; (void)n_in; (void)out_size; (void)d_ws; (void)ws_size;
    mdrnn_scan_kernel<<<dim3(32), dim3(512), 0, stream>>>(x, W_in, b_in, W_state, b_state, out);
}
